// Attention_63737314672745
// MI455X (gfx1250) — matrix operand order repaired and checked by reading, not compiled, not run on hardware
//
#include <hip/hip_runtime.h>
#include <hip/hip_bf16.h>

// ---------------------------------------------------------------------------
// Problem constants (from reference setup_inputs)
// ---------------------------------------------------------------------------
#define BB     4
#define NN     8192
#define DIM    1024
#define HEADS  16
#define DH     64
#define WIN    64
#define NWIN   (NN / WIN)          // 128
#define ROWS   (BB * NN)           // 32768
#define DIM3   (3 * DIM)           // 3072
#define BHD    (BB * HEADS)        // 64

// ---------------------------------------------------------------------------
// bf16 helpers (bit-level, round-to-nearest-even)
// ---------------------------------------------------------------------------
__device__ __forceinline__ unsigned short f2bf(float f) {
    unsigned int u = __float_as_uint(f);
    unsigned int r = u + 0x7FFFu + ((u >> 16) & 1u);
    return (unsigned short)(r >> 16);
}
__device__ __forceinline__ float bf2f(unsigned short h) {
    return __uint_as_float(((unsigned int)h) << 16);
}

// ---------------------------------------------------------------------------
// WMMA fragment types (CDNA5, wave32)
// ---------------------------------------------------------------------------
typedef __attribute__((ext_vector_type(16))) __bf16 v16bf;
typedef __attribute__((ext_vector_type(8)))  float  v8f;

union Frag16 { v16bf v; unsigned int u[8]; };

__device__ __forceinline__ v8f wmma_bf16(const Frag16& a, const Frag16& b, v8f c) {
    // D = A(16x32 bf16) * B(32x16 bf16) + C(16x16 f32)
    return __builtin_amdgcn_wmma_f32_16x16x32_bf16(
        false, a.v, false, b.v, (short)0, c, false, false);
}

// A-fragment (16x32 bf16) from row-major bf16 LDS tile (rowStride in elements).
__device__ __forceinline__ void load_afrag(Frag16& a, const unsigned short* base,
                                           int rowStride, int mBase, int kOff, int lane) {
    int Ml  = lane & 15;
    int hi8 = (lane & 16) ? 8 : 0;
    const unsigned short* rp = base + (mBase + Ml) * rowStride + kOff + hi8;
#pragma unroll
    for (int v = 0; v < 8; ++v) {
        int K0 = ((v & 4) << 2) + ((v & 3) << 1); // v>=4 -> +16
        a.u[v] = *reinterpret_cast<const unsigned int*>(rp + K0);
    }
}

// A-fragment built from an f32 row-major LDS tile (pack pairs to bf16).
__device__ __forceinline__ void load_afrag_f32(Frag16& a, const float* base,
                                               int rowStride, int mBase, int kOff, int lane) {
    int Ml  = lane & 15;
    int hi8 = (lane & 16) ? 8 : 0;
    const float* rp = base + (mBase + Ml) * rowStride + kOff + hi8;
#pragma unroll
    for (int v = 0; v < 8; ++v) {
        int K0 = ((v & 4) << 2) + ((v & 3) << 1);
        unsigned int lo = f2bf(rp[K0]);
        unsigned int hi = f2bf(rp[K0 + 1]);
        a.u[v] = lo | (hi << 16);
    }
}

// B-fragment (32x16 bf16). Storage is N-major / K-contiguous:
// element (k,n) at base[(nBase+n)*rowStride + kOff + k].
__device__ __forceinline__ void load_bfrag(Frag16& b, const unsigned short* base,
                                           int rowStride, int nBase, int kOff, int lane) {
    int Nl   = lane & 15;
    int hi8  = (lane & 16) ? 8 : 0;
    const unsigned short* rp = base + (nBase + Nl) * rowStride + kOff + hi8;
#pragma unroll
    for (int v = 0; v < 8; ++v)
        b.u[v] = *reinterpret_cast<const unsigned int*>(rp + ((v & 4) << 2) + ((v & 3) << 1));
}

// ---------------------------------------------------------------------------
// Tensor Data Mover: 2D tile load (bf16, tile 32 wide along K) per ISA 08 §8.
// g0: count=1 | lds_addr | global_addr | type=2.   g1: data_size=2B, dims.
// ---------------------------------------------------------------------------
#if defined(__gfx1250__) && __has_builtin(__builtin_amdgcn_tensor_load_to_lds) && \
    __has_builtin(__builtin_amdgcn_s_wait_tensorcnt)
#define HAVE_TDM 1
#else
#define HAVE_TDM 0
#endif

typedef __attribute__((ext_vector_type(4))) unsigned int uint32x4;
typedef __attribute__((ext_vector_type(8))) int          int32x8;
typedef __attribute__((ext_vector_type(4))) int          int32x4;

#if HAVE_TDM
__device__ __forceinline__ void tdm_load_tile_2d(const unsigned short* gsrc,
                                                 unsigned ldsOff, int tileRows,
                                                 int strideElems, int tensorRows) {
    unsigned long long ga = (unsigned long long)(size_t)gsrc;
    unsigned td0 = (unsigned)strideElems;   // tensor_dim0 (row length)
    unsigned td1 = (unsigned)tensorRows;    // tensor_dim1
    uint32x4 g0;
    g0[0] = 1u;                                               // count=1, user mode
    g0[1] = ldsOff;                                           // lds_addr (bytes)
    g0[2] = (unsigned)(ga & 0xFFFFFFFFu);                     // global_addr[31:0]
    g0[3] = (unsigned)((ga >> 32) & 0x01FFFFFFu) | 0x80000000u; // addr[56:32]|type=2
    int32x8 g1;
    g1[0] = 0x00010000;                                       // data_size=1 (2 bytes)
    g1[1] = (int)((td0 & 0xFFFFu) << 16);                     // tensor_dim0[15:0]
    g1[2] = (int)((td0 >> 16) | ((td1 & 0xFFFFu) << 16));     // td0 hi | td1 lo
    g1[3] = (int)((td1 >> 16) | (32u << 16));                 // td1 hi | tile_dim0=32
    g1[4] = (int)((unsigned)tileRows & 0xFFFFu);              // tile_dim1 (tile_dim2=0)
    g1[5] = (int)(unsigned)strideElems;                       // tensor_dim0_stride lo32
    g1[6] = 0;                                                // stride hi | dim1_stride lo
    g1[7] = 0;
    int32x4 z4 = {0, 0, 0, 0};
#if defined(__clang_major__) && (__clang_major__ >= 23)
    int32x8 z8 = {0, 0, 0, 0, 0, 0, 0, 0};
    __builtin_amdgcn_tensor_load_to_lds(g0, g1, z4, z4, z8, 0);
#else
    __builtin_amdgcn_tensor_load_to_lds(g0, g1, z4, z4, 0);
#endif
}
#endif

// ---------------------------------------------------------------------------
// Kernel 1: RMSNorm (fp32 in -> bf16 out), one block per row
// ---------------------------------------------------------------------------
__global__ __launch_bounds__(256) void rmsnorm_kernel(const float* __restrict__ x,
                                                      const float* __restrict__ wn,
                                                      unsigned short* __restrict__ xn) {
    __shared__ float red[8];
    int row = blockIdx.x;
    int tid = threadIdx.x;
    const float* xr = x + (size_t)row * DIM;
    float vals[4];
    float ss = 0.f;
#pragma unroll
    for (int e = 0; e < 4; ++e) {
        float f = xr[tid + e * 256];
        vals[e] = f;
        ss += f * f;
    }
#pragma unroll
    for (int off = 16; off > 0; off >>= 1) ss += __shfl_xor(ss, off, 32);
    if ((tid & 31) == 0) red[tid >> 5] = ss;
    __syncthreads();
    float tot = 0.f;
#pragma unroll
    for (int i = 0; i < 8; ++i) tot += red[i];
    float r = rsqrtf(tot * (1.0f / (float)DIM) + 1.1920929e-7f);
    unsigned short* xo = xn + (size_t)row * DIM;
#pragma unroll
    for (int e = 0; e < 4; ++e)
        xo[tid + e * 256] = f2bf(vals[e] * r * wn[tid + e * 256]);
}

// ---------------------------------------------------------------------------
// Kernel 2: f32 (K,Nn) -> bf16 transposed (Nn,K).  Coalesced read.
// ---------------------------------------------------------------------------
__global__ __launch_bounds__(256) void cvtT_kernel(const float* __restrict__ in,
                                                   unsigned short* __restrict__ out,
                                                   int Kdim, int Nn) {
    int i = blockIdx.x * 256 + threadIdx.x;
    if (i < Kdim * Nn) {
        int kk = i / Nn, n = i - kk * Nn;
        out[(size_t)n * Kdim + kk] = f2bf(in[i]);
    }
}

// ---------------------------------------------------------------------------
// Kernel 3: tiled bf16 WMMA GEMM, double-buffered LDS, TDM-staged tiles.
// C = A(M,K) * W^T  with W stored transposed (Nn,K) so both tiles are
// contiguous 2D row-major slices (ideal for tensor_load_to_lds).
// Block tile 128(M) x 64(N), K-step 32. 8 waves = 4(M) x 2(N), 2x2 WMMA each.
// MODE 0: scatter to q/k/v bf16 (b,h,n,dh).   MODE 1: plain f32 store.
// ---------------------------------------------------------------------------
template <int MODE>
__global__ __launch_bounds__(256) void gemm_kernel(
    const unsigned short* __restrict__ A, const unsigned short* __restrict__ Wt,
    int M, int Nn, int K,
    unsigned short* __restrict__ qb, unsigned short* __restrict__ kb,
    unsigned short* __restrict__ vb, float* __restrict__ Cout) {
    __shared__ __align__(16) unsigned short As[2][128 * 32];
    __shared__ __align__(16) unsigned short Bt[2][64 * 32];   // N-major, K-contiguous

    int tid = threadIdx.x;
    int lane = tid & 31;
    int wave = tid >> 5;
    int wm = wave >> 1;            // 0..3, 32 rows each
    int wn = wave & 1;             // 0..1, 32 cols each
    int rowBase = blockIdx.y * 128;
    int colBase = blockIdx.x * 64;

    v8f zero = {};
    v8f acc[2][2];
    acc[0][0] = zero; acc[0][1] = zero; acc[1][0] = zero; acc[1][1] = zero;

    // --- tile staging (TDM: wave 0 issues two descriptors; else vector copy) ---
    auto stage = [&](int k0, int buf) {
#if HAVE_TDM
        if (wave == 0) {
            tdm_load_tile_2d(A  + (size_t)rowBase * K + k0,
                             (unsigned)(size_t)&As[buf][0], 128, K, M);
            tdm_load_tile_2d(Wt + (size_t)colBase * K + k0,
                             (unsigned)(size_t)&Bt[buf][0], 64, K, Nn);
        }
#else
        {   // A tile: thread -> (row=tid/2, 32B half)
            int r = tid >> 1, hh = (tid & 1) << 4;
            const uint4* src = reinterpret_cast<const uint4*>(
                A + (size_t)(rowBase + r) * K + k0 + hh);
            uint4* dst = reinterpret_cast<uint4*>(&As[buf][r * 32 + hh]);
            dst[0] = src[0];
            dst[1] = src[1];
        }
        {   // B tile: thread -> (row=tid/4, 16B quarter)
            int r = tid >> 2, qq = (tid & 3) << 3;
            *reinterpret_cast<uint4*>(&Bt[buf][r * 32 + qq]) =
                *reinterpret_cast<const uint4*>(Wt + (size_t)(colBase + r) * K + k0 + qq);
        }
#endif
    };
    auto stage_wait = [&]() {
#if HAVE_TDM
        if (wave == 0) __builtin_amdgcn_s_wait_tensorcnt(0);
#endif
    };

    int cur = 0;
    stage(0, cur);
    stage_wait();
    __syncthreads();

    for (int k0 = 0; k0 < K; k0 += 32) {
        int nxt = cur ^ 1;
        bool more = (k0 + 32) < K;
        if (more) stage(k0 + 32, nxt);   // DMA next tile while computing this one

        Frag16 af[2], bf[2];
        load_afrag(af[0], As[cur], 32, wm * 32 + 0,  0, lane);
        load_afrag(af[1], As[cur], 32, wm * 32 + 16, 0, lane);
        load_bfrag(bf[0], Bt[cur], 32, wn * 32 + 0,  0, lane);
        load_bfrag(bf[1], Bt[cur], 32, wn * 32 + 16, 0, lane);
        acc[0][0] = wmma_bf16(af[0], bf[0], acc[0][0]);
        acc[0][1] = wmma_bf16(af[0], bf[1], acc[0][1]);
        acc[1][0] = wmma_bf16(af[1], bf[0], acc[1][0]);
        acc[1][1] = wmma_bf16(af[1], bf[1], acc[1][1]);

        if (more) stage_wait();
        __syncthreads();
        cur = nxt;
    }

    int Nl  = lane & 15;
    int hi8 = (lane & 16) ? 8 : 0;
#pragma unroll
    for (int tm = 0; tm < 2; ++tm)
#pragma unroll
        for (int tn = 0; tn < 2; ++tn)
#pragma unroll
            for (int v = 0; v < 8; ++v) {
                int Mg = rowBase + wm * 32 + tm * 16 + v + hi8;
                int Ng = colBase + wn * 32 + tn * 16 + Nl;
                float val = acc[tm][tn][v];
                if (MODE == 0) {
                    int sel  = Ng >> 10;      // 0:q 1:k 2:v
                    int c    = Ng & 1023;
                    int head = c >> 6;
                    int dc   = c & 63;
                    int bi   = Mg >> 13;      // / 8192
                    int pos  = Mg & 8191;
                    size_t dst = ((size_t)(bi * HEADS + head) * NN + pos) * DH + dc;
                    unsigned short* p = (sel == 0) ? qb : (sel == 1) ? kb : vb;
                    p[dst] = f2bf(val);
                } else {
                    Cout[(size_t)Mg * Nn + Ng] = val;
                }
            }
}

// ---------------------------------------------------------------------------
// Kernel 4: RoPE in place on (BH, N, 64) bf16. One thread per (bh,pos,j<32).
// ---------------------------------------------------------------------------
__global__ __launch_bounds__(256) void rope_kernel(unsigned short* __restrict__ t) {
    int g   = blockIdx.x * 256 + threadIdx.x;
    int j   = g & 31;
    int pos = (g >> 5) & (NN - 1);
    int bh  = g >> 18;
    size_t base = ((size_t)bh * NN + pos) * DH;
    float x1 = bf2f(t[base + j]);
    float x2 = bf2f(t[base + j + 32]);
    float inv = __expf(-0.28782313662425572f * (float)j);   // 10000^(-j/32)
    float f = (float)pos * inv;
    float c = __cosf(f), s = __sinf(f);
    t[base + j]      = f2bf(x1 * c - x2 * s);
    t[base + j + 32] = f2bf(x2 * c + x1 * s);
}

// ---------------------------------------------------------------------------
// Kernel 5: sliding-window attention. One block (4 waves) per (b,h,window).
// LDS: Qs 64x64 bf16 | Ks 128x64 bf16 | Vt 64x128 bf16 | S 64x128 f32 (72 KB)
// ---------------------------------------------------------------------------
__global__ __launch_bounds__(128) void attn_kernel(
    const unsigned short* __restrict__ q, const unsigned short* __restrict__ k,
    const unsigned short* __restrict__ v, unsigned short* __restrict__ out) {
    extern __shared__ __align__(16) char smem[];
    unsigned short* Qs = (unsigned short*)smem;   // 64*64
    unsigned short* Ks = Qs + 64 * 64;            // 128*64 (j-major, d contiguous)
    unsigned short* Vt = Ks + 128 * 64;           // 64*128 (d-major, j contiguous)
    float*          S  = (float*)(Vt + 64 * 128); // 64*128 scores/probs

    int tid  = threadIdx.x;
    int lane = tid & 31;
    int wave = tid >> 5;
    int bh   = blockIdx.x >> 7;
    int win  = blockIdx.x & (NWIN - 1);
    int b    = bh >> 4;
    int head = bh & 15;

    const unsigned short* qptr = q + ((size_t)bh * NN + win * WIN) * DH;
    for (int i = tid; i < 64 * 64; i += 128)
        Qs[i] = f2bf(bf2f(qptr[i]) * 0.125f);       // d^-0.5 = 0.125

    {
        int jj = tid;                                // 0..127
        int pos = win * WIN - WIN + jj;
        if (pos >= 0) {
            const uint4* ks4 = (const uint4*)(k + ((size_t)bh * NN + pos) * DH);
            uint4* kd = (uint4*)&Ks[jj * 64];
#pragma unroll
            for (int e = 0; e < 8; ++e) kd[e] = ks4[e];
            const unsigned short* vs = v + ((size_t)bh * NN + pos) * DH;
#pragma unroll
            for (int d = 0; d < 64; ++d) Vt[d * 128 + jj] = vs[d];
        } else {
            uint4 z = {0u, 0u, 0u, 0u};
            uint4* kd = (uint4*)&Ks[jj * 64];
#pragma unroll
            for (int e = 0; e < 8; ++e) kd[e] = z;
#pragma unroll
            for (int d = 0; d < 64; ++d) Vt[d * 128 + jj] = 0;
        }
    }
    __syncthreads();

    int mBase = wave * 16;
    v8f zero = {};

    // Phase 1: S[64x128] = Qs @ Ks^T
    {
        v8f sacc[8];
#pragma unroll
        for (int t = 0; t < 8; ++t) sacc[t] = zero;
#pragma unroll
        for (int kofs = 0; kofs < 64; kofs += 32) {
            Frag16 af;
            load_afrag(af, Qs, 64, mBase, kofs, lane);
#pragma unroll
            for (int tn = 0; tn < 8; ++tn) {
                Frag16 bfr;
                load_bfrag(bfr, Ks, 64, tn * 16, kofs, lane);
                sacc[tn] = wmma_bf16(af, bfr, sacc[tn]);
            }
        }
        int Nl  = lane & 15;
        int hi8 = (lane & 16) ? 8 : 0;
#pragma unroll
        for (int tn = 0; tn < 8; ++tn)
#pragma unroll
            for (int vv = 0; vv < 8; ++vv)
                S[(mBase + vv + hi8) * 128 + tn * 16 + Nl] = sacc[tn][vv];
    }
    __syncthreads();

    // Softmax with causal sliding-window mask (one row per thread)
    if (tid < 64) {
        int i = tid;
        float* row = S + i * 128;
        float mx = -3.402823466e38f;
        for (int j = 0; j < 128; ++j) {
            int jrel = j - WIN;
            int rel  = i - jrel;
            bool ok = (rel >= 0) && (rel <= WIN) && (win * WIN + jrel >= 0);
            float s = ok ? row[j] : -3.402823466e38f;
            row[j] = s;
            mx = fmaxf(mx, s);
        }
        float sum = 0.f;
        for (int j = 0; j < 128; ++j) {
            float e = __expf(row[j] - mx);
            row[j] = e;
            sum += e;
        }
        float inv = 1.0f / sum;
        for (int j = 0; j < 128; ++j) row[j] *= inv;
    }
    __syncthreads();

    // Phase 2: out = P(64x128) @ Vcat(128x64)
    {
        v8f oacc[4];
#pragma unroll
        for (int t = 0; t < 4; ++t) oacc[t] = zero;
#pragma unroll
        for (int kofs = 0; kofs < 128; kofs += 32) {
            Frag16 af;
            load_afrag_f32(af, S, 128, mBase, kofs, lane);
#pragma unroll
            for (int tn = 0; tn < 4; ++tn) {
                Frag16 bfr;
                load_bfrag(bfr, Vt, 128, tn * 16, kofs, lane);
                oacc[tn] = wmma_bf16(af, bfr, oacc[tn]);
            }
        }
        int Nl  = lane & 15;
        int hi8 = (lane & 16) ? 8 : 0;
#pragma unroll
        for (int tn = 0; tn < 4; ++tn)
#pragma unroll
            for (int vv = 0; vv < 8; ++vv) {
                int pos = win * WIN + mBase + vv + hi8;
                int d   = tn * 16 + Nl;
                out[((size_t)b * NN + pos) * DIM + head * DH + d] = f2bf(oacc[tn][vv]);
            }
    }
}

// ---------------------------------------------------------------------------
// Host launcher
// ---------------------------------------------------------------------------
extern "C" void kernel_launch(void* const* d_in, const int* in_sizes, int n_in,
                              void* d_out, int out_size, void* d_ws, size_t ws_size,
                              hipStream_t stream) {
    (void)in_sizes; (void)n_in; (void)out_size; (void)ws_size;
    const float* x      = (const float*)d_in[0];
    const float* w_norm = (const float*)d_in[1];
    const float* w_qkv  = (const float*)d_in[2];
    const float* w_o    = (const float*)d_in[3];

    char* ws = (char*)d_ws;
    unsigned short* xn      = (unsigned short*)ws; ws += (size_t)ROWS * DIM * 2;
    unsigned short* wqkvT   = (unsigned short*)ws; ws += (size_t)DIM * DIM3 * 2; // (3072,1024)
    unsigned short* woT     = (unsigned short*)ws; ws += (size_t)DIM * DIM * 2;  // (1024,1024)
    unsigned short* qb      = (unsigned short*)ws; ws += (size_t)BHD * NN * DH * 2;
    unsigned short* kb      = (unsigned short*)ws; ws += (size_t)BHD * NN * DH * 2;
    unsigned short* vb      = (unsigned short*)ws; ws += (size_t)BHD * NN * DH * 2;
    unsigned short* ao      = (unsigned short*)ws; ws += (size_t)ROWS * DIM * 2;

    // 1) RMSNorm -> bf16
    rmsnorm_kernel<<<ROWS, 256, 0, stream>>>(x, w_norm, xn);
    // 2) weights -> transposed bf16 (N-major, K-contiguous)
    cvtT_kernel<<<(DIM * DIM3) / 256, 256, 0, stream>>>(w_qkv, wqkvT, DIM, DIM3);
    cvtT_kernel<<<(DIM * DIM) / 256, 256, 0, stream>>>(w_o, woT, DIM, DIM);
    // 3) QKV GEMM (TDM-staged tiles, scatter epilogue into q/k/v head layout)
    gemm_kernel<0><<<dim3(DIM3 / 64, ROWS / 128), 256, 0, stream>>>(
        xn, wqkvT, ROWS, DIM3, DIM, qb, kb, vb, nullptr);
    // 4) RoPE on q and k
    rope_kernel<<<(BHD * NN * 32) / 256, 256, 0, stream>>>(qb);
    rope_kernel<<<(BHD * NN * 32) / 256, 256, 0, stream>>>(kb);
    // 5) sliding-window attention (72 KB dynamic LDS)
    attn_kernel<<<BHD * NWIN, 128, 73728, stream>>>(qb, kb, vb, ao);
    // 6) output projection -> fp32 d_out
    gemm_kernel<1><<<dim3(DIM / 64, ROWS / 128), 256, 0, stream>>>(
        ao, woT, ROWS, DIM, DIM, nullptr, nullptr, nullptr, (float*)d_out);
}
